// RAFTFlow_23519240913594
// MI455X (gfx1250) — matrix operand order repaired and checked by reading, not compiled, not run on hardware
//
#include <hip/hip_runtime.h>
#include <math.h>

#define HDIM 96
#define CDIM 64
#define FDIM 128
#define RADIUS 3
#define LEVELS 4
#define NUM_UPDATES 12

typedef __attribute__((ext_vector_type(16))) __bf16 bf16x16;
typedef __attribute__((ext_vector_type(8)))  __bf16 bf16x8;
typedef __attribute__((ext_vector_type(8)))  float  f32x8;

#define ACT_NONE 0
#define ACT_RELU 1
#define ACT_SIGM 2
#define ACT_TANH 3

__device__ __forceinline__ float apply_act(float v, int act) {
  if (act == ACT_RELU) return fmaxf(v, 0.f);
  if (act == ACT_SIGM) return 1.f / (1.f + __expf(-v));
  if (act == ACT_TANH) return tanhf(v);
  return v;
}

// Fragment layouts (ISA 7.12.2, 16-bit A 16x32 / B 32x16):
//   A: lane (l0,hi) row M=l0 holds contiguous K run [hi*16 .. hi*16+15] (order used here)
//   B: lane (l0,hi) col N=l0 holds contiguous K run [hi*16 .. hi*16+15], same order as A (not ISA table order)
//   C/D: element j of lane (l0,hi) is (M = j + hi*8, N = l0)

// ---------------------------------------------------------------------------
// Conv (SAME, stride S) as implicit GEMM on WMMA bf16. Activations are bf16
// NHWC with channel stride padded to 32 (pad channels zero), so the A fragment
// is one aligned 32B load. Each wave computes 16 pixels x 32 output channels
// (2 accumulators; both B fragments come from LDS) so each A fragment feeds
// two WMMAs. Weights for the 32-channel tile staged once per block into LDS
// as bf16 [n(32)][kyx][ci]. Block = 128 threads = 4 waves = 64 pixels.
// ---------------------------------------------------------------------------
template <int K, int S>
__global__ __launch_bounds__(128) void conv2d_wmma_kernel(
    const __bf16* __restrict__ in, const float* __restrict__ wgt,
    const float* __restrict__ bias, __bf16* __restrict__ out,
    int Cin, int Cinpad, int H, int W, int Cout, int Coutpad, int Ho, int Wo, int act)
{
  constexpr int KK = K * K;
  extern __shared__ __align__(16) char smem_raw[];
  __bf16* wlds = (__bf16*)smem_raw;

  const int tid = threadIdx.x;
  const int wave = tid >> 5;
  const int lane = tid & 31;
  const int l0 = lane & 15;
  const int hi = lane >> 4;
  const int co0 = blockIdx.y * 32;
  const int b = blockIdx.z;
  const int HWo = Ho * Wo;
  const int slab = KK * Cinpad;

  // ---- stage bf16 weights for this 32-channel tile into LDS ----
  {
    const int total = 32 * slab;
    for (int i = tid; i < total; i += 128) {
      int n = i / slab;
      int rem = i - n * slab;
      int kyx = rem / Cinpad;
      int ci = rem - kyx * Cinpad;
      float v = 0.f;
      if (co0 + n < Cout && ci < Cin)
        v = wgt[((size_t)(co0 + n) * Cin + ci) * KK + kyx];
      wlds[i] = (__bf16)v;
    }
  }
  __syncthreads();

  const int p0 = (blockIdx.x * 4 + wave) * 16;
  const int m = p0 + l0;
  const int oy = (m < HWo) ? (m / Wo) : 0;
  const int ox = (m < HWo) ? (m % Wo) : 0;
  const int n0 = co0 + l0;
  const int n1 = co0 + 16 + l0;
  int padT = (Ho - 1) * S + K - H;
  if (padT < 0) padT = 0;
  const int padBeg = padT / 2;

  const __bf16* inb = in + (size_t)b * H * W * Cinpad;
  const __bf16* wrow0 = wlds + (size_t)l0 * slab;
  const __bf16* wrow1 = wlds + (size_t)(16 + l0) * slab;

  f32x8 acc0 = {}, acc1 = {};
  for (int kyx = 0; kyx < KK; ++kyx) {
    const int ky = kyx / K;            // compile-time divisor
    const int kx = kyx - ky * K;
    const int iy = oy * S - padBeg + ky;
    const int ix = ox * S - padBeg + kx;
    const bool valid = (m < HWo) && (iy >= 0) && (iy < H) && (ix >= 0) && (ix < W);
    const __bf16* aptr = inb + ((size_t)iy * W + ix) * Cinpad;

    // prefetch next tap's activation row (global_prefetch_b8)
    if (kyx + 1 < KK) {
      const int ky2 = (kyx + 1) / K;
      const int kx2 = (kyx + 1) - ky2 * K;
      const int iy2 = oy * S - padBeg + ky2;
      const int ix2 = ox * S - padBeg + kx2;
      if ((m < HWo) && iy2 >= 0 && iy2 < H && ix2 >= 0 && ix2 < W)
        __builtin_prefetch(inb + ((size_t)iy2 * W + ix2) * Cinpad, 0, 1);
    }

    const __bf16* wk0 = wrow0 + kyx * Cinpad;
    const __bf16* wk1 = wrow1 + kyx * Cinpad;

    for (int ci0 = 0; ci0 < Cinpad; ci0 += 32) {
      bf16x16 av;
      if (valid) {
        av = *(const bf16x16*)(aptr + ci0 + hi * 16);
      } else {
#pragma unroll
        for (int j = 0; j < 16; ++j) av[j] = (__bf16)0.f;
      }
      bf16x16 bv0 = *(const bf16x16*)(wk0 + ci0 + hi * 16);   // 32B ds_loads
      bf16x16 bv1 = *(const bf16x16*)(wk1 + ci0 + hi * 16);
      acc0 = __builtin_amdgcn_wmma_f32_16x16x32_bf16(false, av, false, bv0, (short)0, acc0,
                                                     false, false);
      acc1 = __builtin_amdgcn_wmma_f32_16x16x32_bf16(false, av, false, bv1, (short)0, acc1,
                                                     false, false);
    }
  }

  // Coalesced NHWC store: lanes 0-15 cover 16 consecutive channels of a pixel.
  const float bvv0 = (n0 < Cout && bias) ? bias[n0] : 0.f;
  const float bvv1 = (n1 < Cout && bias) ? bias[n1] : 0.f;
#pragma unroll
  for (int j = 0; j < 8; ++j) {
    int mm = p0 + j + hi * 8;
    if (mm < HWo) {
      size_t px = ((size_t)b * HWo + mm) * Coutpad;
      float v0 = (n0 < Cout) ? apply_act(acc0[j] + bvv0, act) : 0.f;  // zero pad channels
      float v1 = (n1 < Cout) ? apply_act(acc1[j] + bvv1, act) : 0.f;
      out[px + n0] = (__bf16)v0;
      out[px + n1] = (__bf16)v1;
    }
  }
}

// ---------------------------------------------------------------------------
// All-pairs correlation on bf16 NHWC features (stride FDIM=128), 16x64 tile
// per wave: 4 accumulators reuse the A fragment, B fragments are 32B loads.
// ---------------------------------------------------------------------------
__global__ __launch_bounds__(32) void corr_wmma_kernel(
    const __bf16* __restrict__ fa, const __bf16* __restrict__ fb,
    float* __restrict__ corr, int Pn, float scale)
{
  const int lane = threadIdx.x;
  const int l0 = lane & 15;
  const int hi = lane >> 4;
  const int p1 = blockIdx.x * 16;
  const int p2b = blockIdx.y * 64;
  const int b = blockIdx.z;

  const __bf16* arow = fa + ((size_t)b * Pn + p1 + l0) * FDIM;
  const __bf16* bbase = fb + ((size_t)b * Pn + p2b + l0) * FDIM;

  f32x8 acc0 = {}, acc1 = {}, acc2 = {}, acc3 = {};
#pragma unroll
  for (int c0 = 0; c0 < FDIM; c0 += 32) {
    bf16x16 av = *(const bf16x16*)(arow + c0 + hi * 16);

    bf16x16 bv0 = *(const bf16x16*)(bbase + (size_t)0  * FDIM + c0 + hi * 16);
    bf16x16 bv1 = *(const bf16x16*)(bbase + (size_t)16 * FDIM + c0 + hi * 16);
    bf16x16 bv2 = *(const bf16x16*)(bbase + (size_t)32 * FDIM + c0 + hi * 16);
    bf16x16 bv3 = *(const bf16x16*)(bbase + (size_t)48 * FDIM + c0 + hi * 16);
    acc0 = __builtin_amdgcn_wmma_f32_16x16x32_bf16(false, av, false, bv0, (short)0, acc0, false, false);
    acc1 = __builtin_amdgcn_wmma_f32_16x16x32_bf16(false, av, false, bv1, (short)0, acc1, false, false);
    acc2 = __builtin_amdgcn_wmma_f32_16x16x32_bf16(false, av, false, bv2, (short)0, acc2, false, false);
    acc3 = __builtin_amdgcn_wmma_f32_16x16x32_bf16(false, av, false, bv3, (short)0, acc3, false, false);
  }
#pragma unroll
  for (int j = 0; j < 8; ++j) {
    int mm = p1 + j + hi * 8;
    size_t rowoff = ((size_t)b * Pn + mm) * Pn;
    corr[rowoff + p2b + 0  + l0] = acc0[j] * scale;
    corr[rowoff + p2b + 16 + l0] = acc1[j] * scale;
    corr[rowoff + p2b + 32 + l0] = acc2[j] * scale;
    corr[rowoff + p2b + 48 + l0] = acc3[j] * scale;
  }
}

// ---------------------------------------------------------------------------
// Elementwise / data-movement kernels (activations: bf16 NHWC, pad-32 strides)
// ---------------------------------------------------------------------------
__global__ void normalize_nhwc_kernel(const float* __restrict__ x, __bf16* __restrict__ y,
                                      int B, int HW)
{
  int total = B * HW * 32;
  int i = blockIdx.x * blockDim.x + threadIdx.x;
  if (i >= total) return;
  int c = i & 31;
  int p = (i >> 5) % HW;
  int b = i / (HW * 32);
  float v = 0.f;
  if (c < 3) {
    const float meanv[3] = {0.485f, 0.456f, 0.406f};
    const float stdv[3]  = {0.229f, 0.224f, 0.225f};
    float raw = x[((size_t)b * 3 + c) * HW + p] * stdv[c] + meanv[c];
    raw = fminf(fmaxf(raw, 0.f), 1.f);
    v = 2.f * raw - 1.f;
  }
  y[i] = (__bf16)v;
}

__global__ void split_ctx_kernel(const __bf16* __restrict__ ctx, __bf16* __restrict__ hbuf,
                                 __bf16* __restrict__ cbuf, int B, int HW)
{
  const int Ct = HDIM + CDIM;   // 160, multiple of 32
  int total = B * HW * Ct;
  int i = blockIdx.x * blockDim.x + threadIdx.x;
  if (i >= total) return;
  int c = i % Ct;
  int p = (i / Ct) % HW;
  int b = i / (Ct * HW);
  float v = (float)ctx[i];
  size_t pix = (size_t)b * HW + p;
  if (c < HDIM) hbuf[pix * HDIM + c] = (__bf16)tanhf(v);
  else          cbuf[pix * CDIM + (c - HDIM)] = (__bf16)fmaxf(v, 0.f);
}

__global__ void avgpool2_kernel(const float* __restrict__ in, float* __restrict__ out,
                                int N, int Hl, int Wl)
{
  int Ho = Hl / 2, Wo = Wl / 2;
  int total = N * Ho * Wo;
  int i = blockIdx.x * blockDim.x + threadIdx.x;
  if (i >= total) return;
  int x = i % Wo;
  int y = (i / Wo) % Ho;
  int n = i / (Wo * Ho);
  const float* pl = in + (size_t)n * Hl * Wl;
  out[i] = 0.25f * (pl[(2 * y) * Wl + 2 * x] + pl[(2 * y) * Wl + 2 * x + 1] +
                    pl[(2 * y + 1) * Wl + 2 * x] + pl[(2 * y + 1) * Wl + 2 * x + 1]);
}

// corr_feat NHWC bf16 (stride 224): channel lvl*49+k at pixel p
__global__ void corr_lookup_kernel(const float* __restrict__ c0,
                                   const float* __restrict__ c1,
                                   const float* __restrict__ c2,
                                   const float* __restrict__ c3,
                                   const float* __restrict__ flow,
                                   __bf16* __restrict__ out, int B, int h, int w)
{
  int total = B * h * w * LEVELS * 49;
  int i = blockIdx.x * blockDim.x + threadIdx.x;
  if (i >= total) return;
  int kidx = i % 49;
  int lvl = (i / 49) % LEVELS;
  int p = (i / (49 * LEVELS)) % (h * w);
  int b = i / (49 * LEVELS * h * w);
  int x = p % w, y = p / w;
  int HW = h * w;

  float cx = (float)x + flow[((size_t)b * 2 + 0) * HW + p];
  float cy = (float)y + flow[((size_t)b * 2 + 1) * HW + p];
  float dxv = (float)(kidx % 7 - RADIUS);
  float dyv = (float)(kidx / 7 - RADIUS);

  const float* base;
  int Hl, Wl;
  if (lvl == 0)      { base = c0; Hl = h;     Wl = w;     }
  else if (lvl == 1) { base = c1; Hl = h / 2; Wl = w / 2; }
  else if (lvl == 2) { base = c2; Hl = h / 4; Wl = w / 4; }
  else               { base = c3; Hl = h / 8; Wl = w / 8; }
  float scale = (float)(1 << lvl);
  float sx = cx / scale + dxv;
  float sy = cy / scale + dyv;
  const float* plane = base + (size_t)(b * HW + p) * Hl * Wl;

  float x0f = floorf(sx), y0f = floorf(sy);
  float wx = sx - x0f, wy = sy - y0f;
  auto g = [&](float yi, float xi) -> float {
    if (xi < 0.f || xi > (float)(Wl - 1) || yi < 0.f || yi > (float)(Hl - 1)) return 0.f;
    return plane[(int)yi * Wl + (int)xi];
  };
  float v = g(y0f, x0f) * (1.f - wy) * (1.f - wx) +
            g(y0f, x0f + 1.f) * (1.f - wy) * wx +
            g(y0f + 1.f, x0f) * wy * (1.f - wx) +
            g(y0f + 1.f, x0f + 1.f) * wy * wx;
  out[((size_t)b * HW + p) * 224 + (lvl * 49 + kidx)] = (__bf16)v;
}

// NHWC channel concat of up to 4 bf16 sources (u=used ch, st=channel stride);
// destination pad channels are written zero.
__global__ void concat4_nhwc_kernel(const __bf16* __restrict__ s0, int u0, int st0,
                                    const __bf16* __restrict__ s1, int u1, int st1,
                                    const __bf16* __restrict__ s2, int u2, int st2,
                                    const __bf16* __restrict__ s3, int u3, int st3,
                                    __bf16* __restrict__ out, int B, int HW, int Cd)
{
  long long total = (long long)B * HW * Cd;
  long long i = (long long)blockIdx.x * blockDim.x + threadIdx.x;
  if (i >= total) return;
  int c = (int)(i % Cd);
  int p = (int)((i / Cd) % HW);
  int b = (int)(i / ((long long)Cd * HW));
  size_t pix = (size_t)b * HW + p;
  __bf16 v = (__bf16)0.f;
  if (c < u0)                     v = s0[pix * st0 + c];
  else if (c < u0 + u1)           v = s1[pix * st1 + (c - u0)];
  else if (c < u0 + u1 + u2)      v = s2[pix * st2 + (c - u0 - u1)];
  else if (c < u0 + u1 + u2 + u3) v = s3[pix * st3 + (c - u0 - u1 - u2)];
  out[i] = v;
}

__global__ void ewise_mul_bf16_kernel(const __bf16* __restrict__ a,
                                      const __bf16* __restrict__ b,
                                      __bf16* __restrict__ o, int n)
{
  int i = blockIdx.x * blockDim.x + threadIdx.x;
  if (i < n) o[i] = (__bf16)((float)a[i] * (float)b[i]);
}

__global__ void gru_update_bf16_kernel(__bf16* __restrict__ h, const __bf16* __restrict__ z,
                                       const __bf16* __restrict__ q, int n)
{
  int i = blockIdx.x * blockDim.x + threadIdx.x;
  if (i < n) {
    float zv = (float)z[i];
    h[i] = (__bf16)((1.f - zv) * (float)h[i] + zv * (float)q[i]);
  }
}

// pack f32 flow (NCHW, 2ch) -> bf16 NHWC stride 32 (pads zero)
__global__ void flow_pack_kernel(const float* __restrict__ flow, __bf16* __restrict__ out,
                                 int B, int HW)
{
  int total = B * HW * 32;
  int i = blockIdx.x * blockDim.x + threadIdx.x;
  if (i >= total) return;
  int c = i & 31;
  int p = (i >> 5) % HW;
  int b = i / (HW * 32);
  float v = (c < 2) ? flow[((size_t)b * 2 + c) * HW + p] : 0.f;
  out[i] = (__bf16)v;
}

// flow (f32 NCHW) += df (bf16 NHWC stride 32)
__global__ void flow_update_kernel(float* __restrict__ flow, const __bf16* __restrict__ df,
                                   int B, int HW)
{
  int total = B * 2 * HW;
  int i = blockIdx.x * blockDim.x + threadIdx.x;
  if (i >= total) return;
  int p = i % HW;
  int c = (i / HW) % 2;
  int b = i / (2 * HW);
  flow[i] += (float)df[((size_t)b * HW + p) * 32 + c];
}

__global__ void fill_zero_kernel(float* __restrict__ a, int n)
{
  int i = blockIdx.x * blockDim.x + threadIdx.x;
  if (i < n) a[i] = 0.f;
}

__global__ void fill_zero_bf16_kernel(__bf16* __restrict__ a, long long n)
{
  long long i = (long long)blockIdx.x * blockDim.x + threadIdx.x;
  if (i < n) a[i] = (__bf16)0.f;
}

__global__ void resize_bl_kernel(const float* __restrict__ in, float* __restrict__ out,
                                 int NC, int Hin, int Win, int Hout, int Wout, float mult)
{
  int total = NC * Hout * Wout;
  int i = blockIdx.x * blockDim.x + threadIdx.x;
  if (i >= total) return;
  int x = i % Wout;
  int y = (i / Wout) % Hout;
  int n = i / (Wout * Hout);
  float sy = (y + 0.5f) * (float)Hin / (float)Hout - 0.5f;
  float sx = (x + 0.5f) * (float)Win / (float)Wout - 0.5f;
  int y0 = (int)floorf(sy), x0 = (int)floorf(sx);
  float wy = sy - (float)y0, wx = sx - (float)x0;
  int y0c = y0 < 0 ? 0 : (y0 > Hin - 1 ? Hin - 1 : y0);
  int y1c = y0 + 1 < 0 ? 0 : (y0 + 1 > Hin - 1 ? Hin - 1 : y0 + 1);
  int x0c = x0 < 0 ? 0 : (x0 > Win - 1 ? Win - 1 : x0);
  int x1c = x0 + 1 < 0 ? 0 : (x0 + 1 > Win - 1 ? Win - 1 : x0 + 1);
  const float* pl = in + (size_t)n * Hin * Win;
  float v = pl[y0c * Win + x0c] * (1.f - wy) * (1.f - wx) +
            pl[y0c * Win + x1c] * (1.f - wy) * wx +
            pl[y1c * Win + x0c] * wy * (1.f - wx) +
            pl[y1c * Win + x1c] * wy * wx;
  out[i] = v * mult;
}

// ---------------------------------------------------------------------------
// Host orchestration
// ---------------------------------------------------------------------------
static inline void launch_conv(hipStream_t stream, const __bf16* in, int Cin, int H, int W,
                               const float* w, const float* b, int Cout, int K, int S,
                               int act, __bf16* out, int Bn)
{
  int Ho = (H + S - 1) / S, Wo = (W + S - 1) / S;
  int Cinpad = (Cin + 31) & ~31;
  int Coutpad = (Cout + 31) & ~31;
  dim3 grid((unsigned)((Ho * Wo + 63) / 64), (unsigned)(Coutpad / 32), (unsigned)Bn);
  size_t sh = (size_t)32 * K * K * Cinpad * sizeof(__bf16);
  if (K == 7 && S == 2)
    conv2d_wmma_kernel<7, 2><<<grid, 128, sh, stream>>>(in, w, b, out, Cin, Cinpad, H, W, Cout, Coutpad, Ho, Wo, act);
  else if (K == 3 && S == 2)
    conv2d_wmma_kernel<3, 2><<<grid, 128, sh, stream>>>(in, w, b, out, Cin, Cinpad, H, W, Cout, Coutpad, Ho, Wo, act);
  else if (K == 7 && S == 1)
    conv2d_wmma_kernel<7, 1><<<grid, 128, sh, stream>>>(in, w, b, out, Cin, Cinpad, H, W, Cout, Coutpad, Ho, Wo, act);
  else if (K == 3 && S == 1)
    conv2d_wmma_kernel<3, 1><<<grid, 128, sh, stream>>>(in, w, b, out, Cin, Cinpad, H, W, Cout, Coutpad, Ho, Wo, act);
  else
    conv2d_wmma_kernel<1, 1><<<grid, 128, sh, stream>>>(in, w, b, out, Cin, Cinpad, H, W, Cout, Coutpad, Ho, Wo, act);
}

extern "C" void kernel_launch(void* const* d_in, const int* in_sizes, int n_in,
                              void* d_out, int out_size, void* d_ws, size_t ws_size,
                              hipStream_t stream)
{
  (void)in_sizes; (void)n_in; (void)out_size; (void)ws_size;
  const int Bn = 4;
  const int Himg = 448, Wimg = 448;
  const int hh = 56, ww = 56;
  const int Pn = hh * ww;                 // 3136
  const int HWimg = Himg * Wimg;

  const float* frame_t  = (const float*)d_in[0];
  const float* frame_t1 = (const float*)d_in[1];
  auto PW = [&](int i) { return (const float*)d_in[i]; };

  // ---- workspace bump allocator (bytes) ----
  char* wp = (char*)d_ws;
  auto allocB = [&](size_t bytes) -> void* {
    void* r = (void*)wp;
    wp += (bytes + 255) & ~(size_t)255;
    return r;
  };
  auto allocF = [&](size_t n) { return (float*)allocB(n * sizeof(float)); };
  auto allocH = [&](size_t n) { return (__bf16*)allocB(n * sizeof(__bf16)); };

  __bf16* img1  = allocH((size_t)Bn * HWimg * 32);
  __bf16* img2  = allocH((size_t)Bn * HWimg * 32);
  __bf16* t1    = allocH((size_t)Bn * 224 * 224 * 32);
  __bf16* t2    = allocH((size_t)Bn * 112 * 112 * 64);
  __bf16* t3    = allocH((size_t)Bn * Pn * 96);
  __bf16* f1o   = allocH((size_t)Bn * Pn * FDIM);
  __bf16* f2o   = allocH((size_t)Bn * Pn * FDIM);
  __bf16* ctxb  = allocH((size_t)Bn * Pn * (HDIM + CDIM));
  __bf16* hbuf  = allocH((size_t)Bn * Pn * HDIM);
  __bf16* cbuf  = allocH((size_t)Bn * Pn * CDIM);
  float*  corr0 = allocF((size_t)Bn * Pn * Pn);
  float*  corr1 = allocF((size_t)Bn * Pn * 28 * 28);
  float*  corr2 = allocF((size_t)Bn * Pn * 14 * 14);
  float*  corr3 = allocF((size_t)Bn * Pn * 7 * 7);
  float*  flow  = allocF((size_t)Bn * 2 * Pn);
  __bf16* flown = allocH((size_t)Bn * Pn * 32);
  __bf16* corrf = allocH((size_t)Bn * Pn * 224);
  __bf16* mc    = allocH((size_t)Bn * Pn * 96);
  __bf16* mfa   = allocH((size_t)Bn * Pn * 64);
  __bf16* mfb   = allocH((size_t)Bn * Pn * 32);
  __bf16* mcat  = allocH((size_t)Bn * Pn * 128);
  __bf16* mbuf  = allocH((size_t)Bn * Pn * 96);   // 80 used
  __bf16* hx    = allocH((size_t)Bn * Pn * 256);  // 242 used
  __bf16* zbuf  = allocH((size_t)Bn * Pn * HDIM);
  __bf16* rbuf  = allocH((size_t)Bn * Pn * HDIM);
  __bf16* qbuf  = allocH((size_t)Bn * Pn * HDIM);
  __bf16* rh    = allocH((size_t)Bn * Pn * HDIM);
  __bf16* h1o   = allocH((size_t)Bn * Pn * 128);
  __bf16* dfn   = allocH((size_t)Bn * Pn * 32);   // 2 used
  float*  flup  = allocF((size_t)Bn * 2 * HWimg);

  auto ew = [&](long long n) { return dim3((unsigned)((n + 255) / 256)); };
  auto conv = [&](const __bf16* in, int Cin, int H, int W, const float* w, const float* b,
                  int Cout, int K, int S, int act, __bf16* out) {
    launch_conv(stream, in, Cin, H, W, w, b, Cout, K, S, act, out, Bn);
  };

  // ---- input normalization -> bf16 NHWC (3 used / 32 stride) ----
  {
    int tot = Bn * HWimg * 32;
    normalize_nhwc_kernel<<<ew(tot), 256, 0, stream>>>(frame_t, img1, Bn, HWimg);
    normalize_nhwc_kernel<<<ew(tot), 256, 0, stream>>>(frame_t1, img2, Bn, HWimg);
  }

  // ---- feature encoder (img1 -> f1o, img2 -> f2o) ----
  conv(img1, 3, 448, 448, PW(2), PW(3), 32, 7, 2, ACT_RELU, t1);
  conv(t1, 32, 224, 224, PW(4), PW(5), 64, 3, 2, ACT_RELU, t2);
  conv(t2, 64, 112, 112, PW(6), PW(7), 96, 3, 2, ACT_RELU, t3);
  conv(t3, 96, 56, 56, PW(8), PW(9), FDIM, 1, 1, ACT_NONE, f1o);

  conv(img2, 3, 448, 448, PW(2), PW(3), 32, 7, 2, ACT_RELU, t1);
  conv(t1, 32, 224, 224, PW(4), PW(5), 64, 3, 2, ACT_RELU, t2);
  conv(t2, 64, 112, 112, PW(6), PW(7), 96, 3, 2, ACT_RELU, t3);
  conv(t3, 96, 56, 56, PW(8), PW(9), FDIM, 1, 1, ACT_NONE, f2o);

  // ---- context encoder ----
  conv(img1, 3, 448, 448, PW(10), PW(11), 32, 7, 2, ACT_RELU, t1);
  conv(t1, 32, 224, 224, PW(12), PW(13), 64, 3, 2, ACT_RELU, t2);
  conv(t2, 64, 112, 112, PW(14), PW(15), 96, 3, 2, ACT_RELU, t3);
  conv(t3, 96, 56, 56, PW(16), PW(17), HDIM + CDIM, 1, 1, ACT_NONE, ctxb);
  {
    int tot = Bn * Pn * (HDIM + CDIM);
    split_ctx_kernel<<<ew(tot), 256, 0, stream>>>(ctxb, hbuf, cbuf, Bn, Pn);
  }

  // ---- all-pairs correlation volume + pyramid (features already [b][p][c]) ----
  {
    dim3 cg((unsigned)(Pn / 16), (unsigned)(Pn / 64), (unsigned)Bn);
    corr_wmma_kernel<<<cg, dim3(32), 0, stream>>>(f1o, f2o, corr0, Pn,
                                                  1.f / sqrtf((float)FDIM));
  }
  {
    int N = Bn * Pn;
    long long t;
    t = (long long)N * 28 * 28; avgpool2_kernel<<<ew(t), 256, 0, stream>>>(corr0, corr1, N, 56, 56);
    t = (long long)N * 14 * 14; avgpool2_kernel<<<ew(t), 256, 0, stream>>>(corr1, corr2, N, 28, 28);
    t = (long long)N * 7 * 7;   avgpool2_kernel<<<ew(t), 256, 0, stream>>>(corr2, corr3, N, 14, 14);
  }

  // ---- init flow = 0, zero corr_feat pad channels once ----
  {
    int tot = Bn * 2 * Pn;
    fill_zero_kernel<<<ew(tot), 256, 0, stream>>>(flow, tot);
    long long tc = (long long)Bn * Pn * 224;
    fill_zero_bf16_kernel<<<ew(tc), 256, 0, stream>>>(corrf, tc);
  }

  const int nH = Bn * Pn * HDIM;

  // ---- 12 GRU update iterations ----
  for (int it = 0; it < NUM_UPDATES; ++it) {
    {
      int tp = Bn * Pn * 32;
      flow_pack_kernel<<<ew(tp), 256, 0, stream>>>(flow, flown, Bn, Pn);
      int tot = Bn * Pn * LEVELS * 49;
      corr_lookup_kernel<<<ew(tot), 256, 0, stream>>>(corr0, corr1, corr2, corr3, flow,
                                                      corrf, Bn, hh, ww);
    }
    conv(corrf, 196, hh, ww, PW(18), PW(19), 96, 1, 1, ACT_RELU, mc);
    conv(flown, 2, hh, ww, PW(20), PW(21), 64, 7, 1, ACT_RELU, mfa);
    conv(mfa, 64, hh, ww, PW(22), PW(23), 32, 3, 1, ACT_RELU, mfb);
    {
      long long tot = (long long)Bn * Pn * 128;
      concat4_nhwc_kernel<<<ew(tot), 256, 0, stream>>>(mc, 96, 96, mfb, 32, 32,
                                                       mc, 0, 96, mc, 0, 96,
                                                       mcat, Bn, Pn, 128);
    }
    conv(mcat, 128, hh, ww, PW(24), PW(25), 80, 3, 1, ACT_RELU, mbuf);

    // hx = concat(h, m, flow, c): 96+80+2+64 = 242 used / 256 stride
    {
      long long tot = (long long)Bn * Pn * 256;
      concat4_nhwc_kernel<<<ew(tot), 256, 0, stream>>>(hbuf, HDIM, HDIM, mbuf, 80, 96,
                                                       flown, 2, 32, cbuf, CDIM, CDIM,
                                                       hx, Bn, Pn, 256);
    }
    conv(hx, 242, hh, ww, PW(26), PW(27), HDIM, 3, 1, ACT_SIGM, zbuf);
    conv(hx, 242, hh, ww, PW(28), PW(29), HDIM, 3, 1, ACT_SIGM, rbuf);

    ewise_mul_bf16_kernel<<<ew(nH), 256, 0, stream>>>(rbuf, hbuf, rh, nH);
    {
      long long tot = (long long)Bn * Pn * 256;
      concat4_nhwc_kernel<<<ew(tot), 256, 0, stream>>>(rh, HDIM, HDIM, mbuf, 80, 96,
                                                       flown, 2, 32, cbuf, CDIM, CDIM,
                                                       hx, Bn, Pn, 256);
    }
    conv(hx, 242, hh, ww, PW(30), PW(31), HDIM, 3, 1, ACT_TANH, qbuf);

    gru_update_bf16_kernel<<<ew(nH), 256, 0, stream>>>(hbuf, zbuf, qbuf, nH);

    conv(hbuf, HDIM, hh, ww, PW(32), PW(33), 128, 3, 1, ACT_RELU, h1o);
    conv(h1o, 128, hh, ww, PW(34), PW(35), 2, 3, 1, ACT_NONE, dfn);
    {
      int tot = Bn * 2 * Pn;
      flow_update_kernel<<<ew(tot), 256, 0, stream>>>(flow, dfn, Bn, Pn);
    }
  }

  // ---- final: flow_up = 8 * resize(flow, 448x448); u = resize(flow_up, 56x56) / 8 ----
  {
    long long tot = (long long)Bn * 2 * HWimg;
    resize_bl_kernel<<<ew(tot), 256, 0, stream>>>(flow, flup, Bn * 2, hh, ww, Himg, Wimg, 8.f);
    long long tot2 = (long long)Bn * 2 * Pn;
    resize_bl_kernel<<<ew(tot2), 256, 0, stream>>>(flup, (float*)d_out, Bn * 2, Himg, Wimg,
                                                   hh, ww, 0.125f);
  }
}
